// StructuralEncoder_68710886801413
// MI455X (gfx1250) — compile-verified, matrix operand order checked by reading, not run on hardware
//
#include <hip/hip_runtime.h>
#include <hip/hip_bf16.h>
#include <math.h>

// ---------------------------------------------------------------------------
// StructuralEncoder: 2x GATConv + edge MLP + struct loss, for gfx1250 (CDNA5)
// GEMMs: v_wmma_f32_16x16x32_f16, B staged in LDS via Tensor Data Mover,
// A stream double-buffered in registers, B fragments batch-loaded (4 at a
// time) so each group needs a single s_wait_dscnt.
// ---------------------------------------------------------------------------

#define NNODES   50000
#define NEDGES   800000
#define INDIM    256
#define HID      128
#define EH       64
#define NCLS     10
#define FIN      266          // INDIM + NCLS
#define KP1      288          // FIN padded to a multiple of 32
#define NEG_SLOPE 0.2f

typedef __attribute__((ext_vector_type(16))) _Float16     v16h;
typedef __attribute__((ext_vector_type(8)))  _Float16     v8h;
typedef __attribute__((ext_vector_type(8)))  float        v8f;
typedef __attribute__((ext_vector_type(4)))  unsigned int ui32x4;
typedef __attribute__((ext_vector_type(8)))  int          i32x8;
typedef __attribute__((ext_vector_type(4)))  int          i32x4;

// Build a 16x32 f16 WMMA operand fragment from two contiguous 16-byte chunks.
// CDNA5 layout (wave32): lanes 0-15 hold K = {kb..kb+7, kb+16..kb+23} of row
// (lane&15); lanes 16-31 hold K = {kb+8.., kb+24..} — caller passes the two
// chunk pointers already offset for the lane's K base.
__device__ __forceinline__ v16h frag_from_chunks(const _Float16* lo,
                                                 const _Float16* hi) {
  union { struct { v8h lo, hi; } s; v16h v; } u;
  u.s.lo = *(const v8h*)lo;
  u.s.hi = *(const v8h*)hi;
  return u.v;
}

// ---------------------------------------------------------------------------
// TDM: stage a rows x cols f16 tile (row-major, contiguous) into LDS.
// Builds a 2-D Tensor DMA descriptor (D#), groups 0/1 per ISA 8.3/8.4:
//   g0: count=1 | lds_addr | global_addr[56:0] | type=2
//   g1: data_size=1(2B) | tensor_dim0=cols | tensor_dim1=rows |
//       tile_dim0=cols | tile_dim1=rows | tensor_dim0_stride=cols
// This toolchain exposes the 6-arg builtin (extra zero-filled SGPR group).
// Issued by wave 0 only; completion via s_wait_tensorcnt 0 + block barrier.
// ---------------------------------------------------------------------------
#if __has_builtin(__builtin_amdgcn_tensor_load_to_lds)
__device__ __forceinline__ void tdm_load_2d(const void* gptr,
                                            unsigned int lds_off,
                                            int rows, int cols) {
  unsigned long long ga = (unsigned long long)(uintptr_t)gptr;
  ui32x4 g0;
  g0[0] = 1u;                                               // count=1
  g0[1] = lds_off;                                          // lds_addr
  g0[2] = (unsigned int)(ga & 0xFFFFFFFFu);                 // global_addr lo
  g0[3] = (unsigned int)((ga >> 32) & 0x1FFFFFFu) | (2u << 30);  // hi | type=2
  i32x8 g1;
  g1[0] = 1 << 16;                                          // data_size=1 (2B)
  g1[1] = (int)(((unsigned)cols & 0xFFFFu) << 16);          // tensor_dim0 lo16
  g1[2] = (int)(((unsigned)cols >> 16) |
                (((unsigned)rows & 0xFFFFu) << 16));        // dim0 hi | dim1 lo
  g1[3] = (int)(((unsigned)rows >> 16) |
                (((unsigned)cols & 0xFFFFu) << 16));        // dim1 hi | tile_dim0
  g1[4] = rows & 0xFFFF;                                    // tile_dim1 (dim2=0)
  g1[5] = cols;                                             // dim0_stride lo32
  g1[6] = 0;
  g1[7] = 0;
  i32x4 z4 = {0, 0, 0, 0};
  i32x8 z8 = {0, 0, 0, 0, 0, 0, 0, 0};
  __builtin_amdgcn_tensor_load_to_lds(g0, g1, z4, z4, z8, 0);
}
#endif

__device__ __forceinline__ void stage_to_lds(const _Float16* __restrict__ g,
                                             _Float16* l, int rows, int cols) {
#if __has_builtin(__builtin_amdgcn_tensor_load_to_lds)
  if (threadIdx.x < 32) {
    tdm_load_2d(g, (unsigned int)(unsigned long long)(uintptr_t)l, rows, cols);
    __builtin_amdgcn_s_wait_tensorcnt(0);
  }
#else
  const int vecs = rows * cols / 8;
  for (int i = threadIdx.x; i < vecs; i += blockDim.x)
    ((v8h*)l)[i] = ((const v8h*)g)[i];
#endif
  __syncthreads();
}

// ---------------------------------------------------------------------------
// WMMA GEMM:  C[M, NT*16] = A[M, KP](f16) @ B[NT*16, KP](f16)^T
// B staged in LDS (shared by the block's 4 waves); one wave owns a 16-row
// slab x NT column tiles; A double-buffered; B loaded 4 fragments at a time.
// ---------------------------------------------------------------------------
template <int NT, int KP>
__global__ void gemm_wmma_kernel(const _Float16* __restrict__ A,
                                 const _Float16* __restrict__ Bg,
                                 float* __restrict__ C, int M) {
  extern __shared__ _Float16 Blds[];
  stage_to_lds(Bg, Blds, NT * 16, KP);

  const int lane = threadIdx.x & 31;
  const int wave = threadIdx.x >> 5;
  const int slab = blockIdx.x * (blockDim.x >> 5) + wave;
  if (slab * 16 >= M) return;            // after the only barrier: safe
  const int  r   = lane & 15;
  const bool hiL = lane >= 16;
  const int  kofs = hiL ? 8 : 0;

  const _Float16* arow = A + (size_t)(slab * 16 + r) * KP;
  constexpr int KS = KP / 32;
  v8f acc[NT] = {};

  v16h a_cur = frag_from_chunks(arow + kofs, arow + kofs + 16);
#pragma unroll
  for (int s = 0; s < KS; ++s) {
    const int kb = s * 32 + kofs;
    v16h a_nxt = a_cur;
    if (s + 1 < KS)                       // prefetch next A fragment
      a_nxt = frag_from_chunks(arow + kb + 32, arow + kb + 48);
#pragma unroll
    for (int g = 0; g < NT; g += 4) {     // batch 4 B fragments -> 1 DS wait
      v16h bf[4];
#pragma unroll
      for (int j = 0; j < 4; ++j) {
        const _Float16* brow = Blds + (size_t)((g + j) * 16 + r) * KP;
        bf[j] = frag_from_chunks(brow + kb, brow + kb + 16);
      }
#pragma unroll
      for (int j = 0; j < 4; ++j)
        acc[g + j] = __builtin_amdgcn_wmma_f32_16x16x32_f16(
            false, a_cur, false, bf[j], (short)0, acc[g + j], false, false);
    }
    a_cur = a_nxt;
  }
  // D layout: element i -> row slab*16 + i + (hiL?8:0), col t*16 + (lane&15)
  const int ldc   = NT * 16;
  const int rbase = slab * 16 + (hiL ? 8 : 0);
#pragma unroll
  for (int t = 0; t < NT; ++t)
#pragma unroll
    for (int i = 0; i < 8; ++i)
      C[(size_t)(rbase + i) * ldc + t * 16 + r] = acc[t][i];
}

// ---------------------------------------------------------------------------
// Elementwise / setup kernels
// ---------------------------------------------------------------------------
__global__ void convert_pad_f16_kernel(const float* __restrict__ src,
                                       _Float16* __restrict__ dst,
                                       int rows, int cin, int cpad) {
  int i = blockIdx.x * blockDim.x + threadIdx.x;
  if (i >= rows * cpad) return;
  int rr = i / cpad, c = i - rr * cpad;
  dst[i] = (c < cin) ? (_Float16)src[(size_t)rr * cin + c] : (_Float16)0.0f;
}

__global__ void build_xin_kernel(const float* __restrict__ x,
                                 const int* __restrict__ y,
                                 const unsigned char* __restrict__ mask,
                                 _Float16* __restrict__ xin) {
  int i = blockIdx.x * blockDim.x + threadIdx.x;
  if (i >= NNODES * KP1) return;
  int n = i / KP1, c = i - n * KP1;
  float v = 0.0f;
  if (c < INDIM)      v = x[(size_t)n * INDIM + c];
  else if (c < FIN)   v = (mask[n] && y[n] == (c - INDIM)) ? 1.0f : 0.0f;
  xin[i] = (_Float16)v;
}

__global__ void zero_f32_kernel(float* __restrict__ p, int n) {
  int i = blockIdx.x * blockDim.x + threadIdx.x;
  if (i < n) p[i] = 0.0f;
}

__global__ void init_nodes_kernel(float* __restrict__ mmax,
                                  float* __restrict__ den) {
  int n = blockIdx.x * blockDim.x + threadIdx.x;
  if (n >= NNODES) return;
  mmax[n] = -__builtin_inff();
  den[n]  = 0.0f;
}

__global__ void alpha_kernel(const float* __restrict__ h,
                             const float* __restrict__ a_s,
                             const float* __restrict__ a_d,
                             float* __restrict__ as_, float* __restrict__ ad_) {
  int n = blockIdx.x * blockDim.x + threadIdx.x;
  if (n >= NNODES) return;
  const float* row = h + (size_t)n * HID;
  float s = 0.0f, d = 0.0f;
#pragma unroll 8
  for (int k = 0; k < HID; ++k) { s += row[k] * a_s[k]; d += row[k] * a_d[k]; }
  as_[n] = s; ad_[n] = d;
}

// Monotone-key float atomic max (init to -inf): nonneg via int max, neg via
// uint min — correct for mixed signs.
__device__ __forceinline__ void atomicMaxFloat(float* addr, float v) {
  if (v >= 0.0f) atomicMax((int*)addr, __float_as_int(v));
  else           atomicMin((unsigned int*)addr, __float_as_uint(v));
}

__device__ __forceinline__ void edge_sd(const int* __restrict__ ei, int i,
                                        int& s, int& d) {
  if (i < NEDGES) { s = ei[i]; d = ei[NEDGES + i]; }
  else            { s = d = i - NEDGES; }           // self loops
}

__global__ void att_max_kernel(const int* __restrict__ ei,
                               const float* __restrict__ as_,
                               const float* __restrict__ ad_,
                               float* __restrict__ mmax, int T) {
  int i = blockIdx.x * blockDim.x + threadIdx.x;
  if (i >= T) return;
  int s, d; edge_sd(ei, i, s, d);
  float e = as_[s] + ad_[d];
  e = e > 0.0f ? e : NEG_SLOPE * e;
  atomicMaxFloat(&mmax[d], e);
}

__global__ void att_den_kernel(const int* __restrict__ ei,
                               const float* __restrict__ as_,
                               const float* __restrict__ ad_,
                               const float* __restrict__ mmax,
                               float* __restrict__ den, int T) {
  int i = blockIdx.x * blockDim.x + threadIdx.x;
  if (i >= T) return;
  int s, d; edge_sd(ei, i, s, d);
  float e = as_[s] + ad_[d];
  e = e > 0.0f ? e : NEG_SLOPE * e;
  atomicAdd(&den[d], __expf(e - mmax[d]));
}

// One wave per edge item: 128 channels / 32 lanes = 4 vector atomics per lane.
__global__ void att_agg_kernel(const int* __restrict__ ei,
                               const float* __restrict__ as_,
                               const float* __restrict__ ad_,
                               const float* __restrict__ mmax,
                               const float* __restrict__ den,
                               const float* __restrict__ h,
                               float* __restrict__ aggr, int T) {
  const int lane = threadIdx.x & 31;
  const int i    = blockIdx.x * (blockDim.x >> 5) + (threadIdx.x >> 5);
  if (i >= T) return;
  int s, d; edge_sd(ei, i, s, d);
  float e = as_[s] + ad_[d];
  e = e > 0.0f ? e : NEG_SLOPE * e;
  const float coef = __expf(e - mmax[d]) / den[d];
  const float* hs = h    + (size_t)s * HID;
  float*       od = aggr + (size_t)d * HID;
#pragma unroll
  for (int c = lane; c < HID; c += 32) atomicAdd(&od[c], coef * hs[c]);
}

__global__ void relu_bias_f16_kernel(const float* __restrict__ aggr,
                                     const float* __restrict__ b,
                                     _Float16* __restrict__ out) {
  int i = blockIdx.x * blockDim.x + threadIdx.x;
  if (i >= NNODES * HID) return;
  float v = aggr[i] + b[i & (HID - 1)];
  out[i] = (_Float16)(v > 0.0f ? v : 0.0f);
}

__global__ void bias_f16_kernel(const float* __restrict__ aggr,
                                const float* __restrict__ b,
                                _Float16* __restrict__ out) {
  int i = blockIdx.x * blockDim.x + threadIdx.x;
  if (i >= NNODES * HID) return;
  out[i] = (_Float16)(aggr[i] + b[i & (HID - 1)]);
}

// ---------------------------------------------------------------------------
// Edge MLP: gather-GEMM [16 edges x 256] @ Wm1^T via WMMA (Wm1 TDM-staged in
// LDS, 4 B fragments batch-loaded per K-step), then fused 64->3 linear +
// softmax + loss partials.  4 waves/block.
// Dynamic LDS: [ Wm1 f16 64x256 | hid f32 4x16x65 ]
// ---------------------------------------------------------------------------
__global__ void edge_mlp_kernel(const _Float16* __restrict__ h16,
                                const int* __restrict__ ei,
                                const _Float16* __restrict__ Wm1g,
                                const float* __restrict__ bm1,
                                const float* __restrict__ Wm2,
                                const float* __restrict__ bm2,
                                float* __restrict__ logits,
                                float* __restrict__ probs,
                                float* __restrict__ loss_acc) {
  extern __shared__ _Float16 smem[];
  _Float16* Wm1 = smem;                                   // 64*256 f16
  float*    hid = (float*)(smem + EH * 2 * HID);          // 4*16*65 f32
  stage_to_lds(Wm1g, Wm1, EH, 2 * HID);

  const int  lane = threadIdx.x & 31;
  const int  wave = threadIdx.x >> 5;
  const int  tile = blockIdx.x * 4 + wave;      // 50000 tiles, exact
  const int  r    = lane & 15;
  const bool hiL  = lane >= 16;

  const int e  = tile * 16 + r;
  const int sn = ei[e];
  const int dn = ei[NEDGES + e];
  const _Float16* hs = h16 + (size_t)sn * HID;
  const _Float16* hd = h16 + (size_t)dn * HID;

  v8f acc[4] = {};
#pragma unroll
  for (int s = 0; s < 8; ++s) {                 // K = 256 = 8 x 32
    const int kb = s * 32 + (hiL ? 8 : 0);
    const int k2 = kb + 16;
    // 8-element chunks never straddle the src|dst boundary (128 % 8 == 0)
    const _Float16* plo = (kb < HID) ? (hs + kb) : (hd + kb - HID);
    const _Float16* phi = (k2 < HID) ? (hs + k2) : (hd + k2 - HID);
    v16h a = frag_from_chunks(plo, phi);
    v16h bf[4];
#pragma unroll
    for (int t = 0; t < 4; ++t) {               // batch 4 B frags -> 1 wait
      const _Float16* brow = Wm1 + (size_t)(t * 16 + r) * 256;
      bf[t] = frag_from_chunks(brow + kb, brow + kb + 16);
    }
#pragma unroll
    for (int t = 0; t < 4; ++t)
      acc[t] = __builtin_amdgcn_wmma_f32_16x16x32_f16(
          false, a, false, bf[t], (short)0, acc[t], false, false);
  }
  // bias + ReLU -> LDS (row = local edge, col = hidden unit; pitch 65)
#pragma unroll
  for (int t = 0; t < 4; ++t) {
    const int col = t * 16 + r;
    const float bb = bm1[col];
#pragma unroll
    for (int i = 0; i < 8; ++i) {
      float v = acc[t][i] + bb;
      hid[(wave * 16 + i + (hiL ? 8 : 0)) * (EH + 1) + col] = v > 0.0f ? v : 0.0f;
    }
  }
  __syncthreads();

  if (lane < 16) {
    const int eg = tile * 16 + lane;
    const float* hv = hid + (wave * 16 + lane) * (EH + 1);
    float l0 = bm2[0], l1 = bm2[1], l2 = bm2[2];
#pragma unroll
    for (int k = 0; k < EH; ++k) {
      const float v = hv[k];
      l0 += v * Wm2[k];
      l1 += v * Wm2[EH + k];
      l2 += v * Wm2[2 * EH + k];
    }
    logits[eg * 3 + 0] = l0; logits[eg * 3 + 1] = l1; logits[eg * 3 + 2] = l2;
    const float m  = fmaxf(l0, fmaxf(l1, l2));
    const float e0 = __expf(l0 - m), e1 = __expf(l1 - m), e2 = __expf(l2 - m);
    const float Z  = e0 + e1 + e2;
    const float p0 = e0 / Z, p1 = e1 / Z, p2 = e2 / Z;
    probs[eg * 3 + 0] = p0; probs[eg * 3 + 1] = p1; probs[eg * 3 + 2] = p2;
    const float prior = logf(1.0f / 3.0f + 1e-12f);
    const float kl = p0 * (logf(fmaxf(p0, 1e-12f)) - prior)
                   + p1 * (logf(fmaxf(p1, 1e-12f)) - prior)
                   + p2 * (logf(fmaxf(p2, 1e-12f)) - prior);
    const float rec = logf(fmaxf(p0 + p2, 1e-12f));
    atomicAdd(&loss_acc[0], kl);
    atomicAdd(&loss_acc[1], rec);
  }
}

__global__ void finalize_loss_kernel(const float* __restrict__ acc,
                                     float* __restrict__ out) {
  out[0] = (acc[0] - acc[1]) * (1.0f / (float)NEDGES);
}

// ---------------------------------------------------------------------------
// Host launcher
// ---------------------------------------------------------------------------
extern "C" void kernel_launch(void* const* d_in, const int* in_sizes, int n_in,
                              void* d_out, int out_size, void* d_ws,
                              size_t ws_size, hipStream_t stream) {
  const float*         x      = (const float*)d_in[0];
  const int*           ei     = (const int*)d_in[1];
  const int*           y      = (const int*)d_in[2];
  const unsigned char* mask   = (const unsigned char*)d_in[3];
  const float*         W1     = (const float*)d_in[4];
  const float*         a_src1 = (const float*)d_in[5];
  const float*         a_dst1 = (const float*)d_in[6];
  const float*         b1     = (const float*)d_in[7];
  const float*         W2     = (const float*)d_in[8];
  const float*         a_src2 = (const float*)d_in[9];
  const float*         a_dst2 = (const float*)d_in[10];
  const float*         b2     = (const float*)d_in[11];
  const float*         Wm1    = (const float*)d_in[12];
  const float*         bm1    = (const float*)d_in[13];
  const float*         Wm2    = (const float*)d_in[14];
  const float*         bm2    = (const float*)d_in[15];
  float* out = (float*)d_out;

  // workspace carve-up (256B aligned)
  char*  ws = (char*)d_ws;
  size_t o  = 0;
  auto carve = [&](size_t bytes) {
    size_t r = o;
    o = (o + bytes + 255) & ~(size_t)255;
    return r;
  };
  _Float16* xin16  = (_Float16*)(ws + carve((size_t)NNODES * KP1 * 2));
  _Float16* w1f16  = (_Float16*)(ws + carve((size_t)HID * KP1 * 2));
  _Float16* w2f16  = (_Float16*)(ws + carve((size_t)HID * HID * 2));
  _Float16* wm1f16 = (_Float16*)(ws + carve((size_t)EH * 2 * HID * 2));
  float*    hbuf   = (float*)   (ws + carve((size_t)NNODES * HID * 4)); // hA/hB
  float*    aggr   = (float*)   (ws + carve((size_t)NNODES * HID * 4));
  _Float16* z16    = (_Float16*)(ws + carve((size_t)NNODES * HID * 2)); // z / h_final
  float*    as_    = (float*)   (ws + carve((size_t)NNODES * 4));
  float*    ad_    = (float*)   (ws + carve((size_t)NNODES * 4));
  float*    mx_    = (float*)   (ws + carve((size_t)NNODES * 4));
  float*    dn_    = (float*)   (ws + carve((size_t)NNODES * 4));
  float*    loss   = (float*)   (ws + carve(2 * 4));
  (void)ws_size; (void)n_in; (void)in_sizes; (void)out_size;

  const int TPB = 256;
  auto cdiv = [](int a, int b) { return (a + b - 1) / b; };
  const int T = NEDGES + NNODES;          // edge list incl. self loops

  // --- weight conversion + input assembly ---------------------------------
  convert_pad_f16_kernel<<<cdiv(HID * KP1, TPB), TPB, 0, stream>>>(W1, w1f16, HID, FIN, KP1);
  convert_pad_f16_kernel<<<cdiv(HID * HID, TPB), TPB, 0, stream>>>(W2, w2f16, HID, HID, HID);
  convert_pad_f16_kernel<<<cdiv(EH * 2 * HID, TPB), TPB, 0, stream>>>(Wm1, wm1f16, EH, 2 * HID, 2 * HID);
  build_xin_kernel<<<cdiv(NNODES * KP1, TPB), TPB, 0, stream>>>(x, y, mask, xin16);

  const int slabs     = NNODES / 16;      // 3125
  const int gemm_grid = cdiv(slabs, 4);   // 4 waves / block
  const size_t lds1   = (size_t)HID * KP1 * 2;                  // 73728 B
  const size_t lds2   = (size_t)HID * HID * 2;                  // 32768 B
  const size_t ldsE   = (size_t)EH * 2 * HID * 2
                      + (size_t)4 * 16 * (EH + 1) * 4;          // 49408 B

  // --- GAT layer 1 --------------------------------------------------------
  gemm_wmma_kernel<8, KP1><<<gemm_grid, 128, lds1, stream>>>(xin16, w1f16, hbuf, NNODES);
  alpha_kernel<<<cdiv(NNODES, TPB), TPB, 0, stream>>>(hbuf, a_src1, a_dst1, as_, ad_);
  init_nodes_kernel<<<cdiv(NNODES, TPB), TPB, 0, stream>>>(mx_, dn_);
  zero_f32_kernel<<<cdiv(NNODES * HID, TPB), TPB, 0, stream>>>(aggr, NNODES * HID);
  att_max_kernel<<<cdiv(T, TPB), TPB, 0, stream>>>(ei, as_, ad_, mx_, T);
  att_den_kernel<<<cdiv(T, TPB), TPB, 0, stream>>>(ei, as_, ad_, mx_, dn_, T);
  att_agg_kernel<<<cdiv(T, 8), TPB, 0, stream>>>(ei, as_, ad_, mx_, dn_, hbuf, aggr, T);
  relu_bias_f16_kernel<<<cdiv(NNODES * HID, TPB), TPB, 0, stream>>>(aggr, b1, z16);

  // --- GAT layer 2 --------------------------------------------------------
  gemm_wmma_kernel<8, HID><<<gemm_grid, 128, lds2, stream>>>(z16, w2f16, hbuf, NNODES);
  alpha_kernel<<<cdiv(NNODES, TPB), TPB, 0, stream>>>(hbuf, a_src2, a_dst2, as_, ad_);
  init_nodes_kernel<<<cdiv(NNODES, TPB), TPB, 0, stream>>>(mx_, dn_);
  zero_f32_kernel<<<cdiv(NNODES * HID, TPB), TPB, 0, stream>>>(aggr, NNODES * HID);
  att_max_kernel<<<cdiv(T, TPB), TPB, 0, stream>>>(ei, as_, ad_, mx_, T);
  att_den_kernel<<<cdiv(T, TPB), TPB, 0, stream>>>(ei, as_, ad_, mx_, dn_, T);
  att_agg_kernel<<<cdiv(T, 8), TPB, 0, stream>>>(ei, as_, ad_, mx_, dn_, hbuf, aggr, T);
  bias_f16_kernel<<<cdiv(NNODES * HID, TPB), TPB, 0, stream>>>(aggr, b2, z16); // h_final f16

  // --- edge MLP + loss ----------------------------------------------------
  zero_f32_kernel<<<1, 32, 0, stream>>>(loss, 2);
  edge_mlp_kernel<<<NEDGES / 16 / 4, 128, ldsE, stream>>>(
      z16, ei, wm1f16, bm1, Wm2, bm2,
      out,                         // edge_logits  [E,3]
      out + (size_t)3 * NEDGES,    // edge_probs   [E,3]
      loss);
  finalize_loss_kernel<<<1, 1, 0, stream>>>(loss, out + (size_t)6 * NEDGES);
}
